// MambaBlock_6373731467396
// MI455X (gfx1250) — compile-verified, matrix operand order checked by reading, not run on hardware
//
#include <hip/hip_runtime.h>
#include <hip/hip_bf16.h>
#include <math.h>

typedef __attribute__((ext_vector_type(16))) _Float16 v16h;
typedef __attribute__((ext_vector_type(8)))  float    v8f;
typedef unsigned int u32x4 __attribute__((ext_vector_type(4)));
typedef int          i32x4 __attribute__((ext_vector_type(4)));
typedef int          i32x8 __attribute__((ext_vector_type(8)));

#define BATCH   4
#define DMODEL  64
#define DSTATE  16
#define DINNER  128
#define NSL     16
#define LSEQ    4096          // 64*64
#define BL      (BATCH*LSEQ)  // 16384
#define NPAD    48            // 36 -> 3 WMMA tiles
#define CHUNK   8             // scan steps per TDM tile

// ---------------------------------------------------------------- helpers
__device__ __forceinline__ int smap(int t, int mode) {
  if (mode == 0) return t;                          // forward
  if (mode == 1) return LSEQ - 1 - t;               // flipped
  return (t & (NSL - 1)) * (LSEQ / NSL) + (t >> 4); // slice-interleave
}
__device__ __forceinline__ float gelu_exact(float x) {
  return 0.5f * x * (1.0f + erff(x * 0.70710678118654752f));
}
__device__ __forceinline__ float silu(float x) { return x / (1.0f + __expf(-x)); }

// TDM: load nrows x NPAD f32 tile (row stride NPAD) from global into LDS.
// D# packed per CDNA5 ISA ch.8 (group0: count/lds/global/type, group1: dims).
__device__ __forceinline__ void tdm_load_rows(const float* gsrc, void* lds_dst, int nrows) {
  unsigned lds_off = (unsigned)(unsigned long long)(uintptr_t)lds_dst; // low32 of LDS aperture = LDS byte addr
  unsigned long long ga = (unsigned long long)(uintptr_t)gsrc;
  u32x4 g0;
  g0[0] = 1u;                                        // count=1 (user descriptor)
  g0[1] = lds_off;                                   // lds_addr
  g0[2] = (unsigned)(ga & 0xFFFFFFFFu);              // global_addr[31:0]
  g0[3] = (unsigned)((ga >> 32) & 0x01FFFFFFu) | 0x80000000u; // addr[56:32] | type=2
  i32x8 g1;
  g1[0] = 0x00020000;                                // data_size=4B, no padding/iterate
  g1[1] = (NPAD & 0xFFFF) << 16;                     // tensor_dim0 = 48
  g1[2] = (int)(((unsigned)LSEQ & 0xFFFFu) << 16);   // tensor_dim1 = 4096
  g1[3] = (NPAD & 0xFFFF) << 16;                     // tile_dim0 = 48
  g1[4] = nrows & 0xFFFF;                            // tile_dim1 = CHUNK, tile_dim2=0
  g1[5] = NPAD;                                      // tensor_dim0_stride = 48
  g1[6] = 0;
  g1[7] = 0;
  i32x4 z4 = {0, 0, 0, 0};
#if __clang_major__ >= 23
  i32x8 z8 = {0, 0, 0, 0, 0, 0, 0, 0};
  __builtin_amdgcn_tensor_load_to_lds(g0, g1, z4, z4, z8, 0);
#else
  __builtin_amdgcn_tensor_load_to_lds(g0, g1, z4, z4, 0);
#endif
}

// ---------------------------------------------------------------- small kernels
__global__ void cvt_f16(const float* __restrict__ s, _Float16* __restrict__ d, int n) {
  int i = blockIdx.x * blockDim.x + threadIdx.x;
  if (i < n) d[i] = (_Float16)s[i];
}
__global__ void zero_f16(_Float16* __restrict__ d, int n) {
  int i = blockIdx.x * blockDim.x + threadIdx.x;
  if (i < n) d[i] = (_Float16)0.0f;
}

// LayerNorm over C=64 (wave32 per token), NCHW gather variant + row-major variant
__global__ void ln_nchw(const float* __restrict__ x, const float* __restrict__ g,
                        const float* __restrict__ be, _Float16* __restrict__ out) {
  int warp = blockIdx.x * (blockDim.x >> 5) + (threadIdx.x >> 5);
  if (warp >= BL) return;
  int lane = threadIdx.x & 31;
  int b = warp >> 12, l = warp & (LSEQ - 1);
  const float* base = x + (size_t)b * DMODEL * LSEQ + l;
  float v0 = base[(size_t)lane * LSEQ];
  float v1 = base[(size_t)(lane + 32) * LSEQ];
  float s = v0 + v1;
  #pragma unroll
  for (int o = 16; o > 0; o >>= 1) s += __shfl_xor(s, o, 32);
  float mu = s * (1.0f / 64.0f);
  float d0 = v0 - mu, d1 = v1 - mu;
  float q = d0 * d0 + d1 * d1;
  #pragma unroll
  for (int o = 16; o > 0; o >>= 1) q += __shfl_xor(q, o, 32);
  float rs = rsqrtf(q * (1.0f / 64.0f) + 1e-5f);
  _Float16* orow = out + (size_t)warp * 64;
  orow[lane]      = (_Float16)(d0 * rs * g[lane]      + be[lane]);
  orow[lane + 32] = (_Float16)(d1 * rs * g[lane + 32] + be[lane + 32]);
}
__global__ void ln_rowmajor(const float* __restrict__ x, const float* __restrict__ g,
                            const float* __restrict__ be, _Float16* __restrict__ out) {
  int warp = blockIdx.x * (blockDim.x >> 5) + (threadIdx.x >> 5);
  if (warp >= BL) return;
  int lane = threadIdx.x & 31;
  const float* base = x + (size_t)warp * 64;
  float v0 = base[lane], v1 = base[lane + 32];
  float s = v0 + v1;
  #pragma unroll
  for (int o = 16; o > 0; o >>= 1) s += __shfl_xor(s, o, 32);
  float mu = s * (1.0f / 64.0f);
  float d0 = v0 - mu, d1 = v1 - mu;
  float q = d0 * d0 + d1 * d1;
  #pragma unroll
  for (int o = 16; o > 0; o >>= 1) q += __shfl_xor(q, o, 32);
  float rs = rsqrtf(q * (1.0f / 64.0f) + 1e-5f);
  _Float16* orow = out + (size_t)warp * 64;
  orow[lane]      = (_Float16)(d0 * rs * g[lane]      + be[lane]);
  orow[lane + 32] = (_Float16)(d1 * rs * g[lane + 32] + be[lane + 32]);
}

// ---------------------------------------------------------------- WMMA GEMM
// C[M,N] = A[M,K] * Bw[N,K]^T (f16 in, f32 acc). One wave owns one 16-row
// M-tile and NT consecutive N-tiles: A fragment loaded once per K-step and
// reused across NT v_wmma_f32_16x16x32_f16 issues.
template <int EPI, int NT>
__global__ void gemm_wmma(const _Float16* __restrict__ A, const _Float16* __restrict__ Bw,
                          int M, int N, int K,
                          float* __restrict__ out, const float* __restrict__ extra,
                          _Float16* __restrict__ outh) {
  int wave = blockIdx.x * (blockDim.x >> 5) + (threadIdx.x >> 5);
  int groupsN = (N >> 4) / NT;
  int tm = wave / groupsN, tg = wave - tm * groupsN;
  if (tm >= (M >> 4)) return;
  int lane = threadIdx.x & 31;
  int hi   = lane >> 4;     // half-wave K+8 shift
  int r    = lane & 15;     // row (A) / column (B) within tile

  const _Float16* arow = A + (size_t)((tm << 4) + r) * K + hi * 8;
  const _Float16* brow[NT];
  #pragma unroll
  for (int j = 0; j < NT; ++j)
    brow[j] = Bw + (size_t)(((tg * NT + j) << 4) + r) * K + hi * 8;

  v8f acc[NT];
  #pragma unroll
  for (int j = 0; j < NT; ++j) acc[j] = (v8f){0.f,0.f,0.f,0.f,0.f,0.f,0.f,0.f};

  for (int k0 = 0; k0 < K; k0 += 32) {
    v16h af;
    #pragma unroll
    for (int v = 0; v < 8; ++v) {
      int kk = k0 + ((v & 3) << 1) + ((v >> 2) << 4);
      af[2 * v] = arow[kk]; af[2 * v + 1] = arow[kk + 1];
    }
    #pragma unroll
    for (int j = 0; j < NT; ++j) {
      v16h bf;
      #pragma unroll
      for (int v = 0; v < 8; ++v) {
        int kk = k0 + ((v & 3) << 1) + ((v >> 2) << 4);
        bf[2 * v] = brow[j][kk]; bf[2 * v + 1] = brow[j][kk + 1];
      }
      acc[j] = __builtin_amdgcn_wmma_f32_16x16x32_f16(false, af, false, bf,
                                                      (short)0, acc[j], false, false);
    }
  }

  int mb = (tm << 4) + hi * 8;
  #pragma unroll
  for (int j = 0; j < NT; ++j) {
    int n = ((tg * NT + j) << 4) + (lane & 15);
    #pragma unroll
    for (int v = 0; v < 8; ++v) {
      int m = mb + v;
      float val = acc[j][v];
      int bb = m >> 12, ll = m & (LSEQ - 1);
      if (EPI == 0) {        // in_proj -> xz (b,256,l) f16
        outh[((size_t)bb * 256 + n) * LSEQ + ll] = (_Float16)val;
      } else if (EPI == 1) { // x_proj -> (BL x NPAD) f32
        out[(size_t)m * N + n] = val;
      } else if (EPI == 2) { // out_proj + NCHW residual -> h2 (BL x 64)
        out[(size_t)m * 64 + n] = val + extra[((size_t)bb * 64 + n) * LSEQ + ll];
      } else if (EPI == 3) { // ff1 + gelu -> (B,256,H,W)
        out[((size_t)bb * 256 + n) * LSEQ + ll] = gelu_exact(val);
      } else {               // ff2 + residual -> NCHW output
        out[((size_t)bb * 64 + n) * LSEQ + ll] = val + extra[(size_t)m * 64 + n];
      }
    }
  }
}

// ---------------------------------------------------------------- branch kernels
__global__ void conv_silu(const _Float16* __restrict__ xz, const float* __restrict__ cw,
                          const float* __restrict__ cb, float* __restrict__ xc32,
                          _Float16* __restrict__ xc16, int mode) {
  int idx = blockIdx.x * blockDim.x + threadIdx.x;  // B*128*L
  if (idx >= BATCH * DINNER * LSEQ) return;
  int t = idx & (LSEQ - 1);
  int d = (idx >> 12) & (DINNER - 1);
  int b = idx >> 19;
  const _Float16* xrow = xz + ((size_t)b * 256 + d) * LSEQ;
  float acc = cb[d];
  #pragma unroll
  for (int k = 0; k < 4; ++k) {
    int ts = t - 3 + k;
    if (ts >= 0) acc += cw[d * 4 + k] * (float)xrow[smap(ts, mode)];
  }
  float sv = silu(acc);
  xc32[((size_t)b * DINNER + d) * LSEQ + t] = sv;
  xc16[((size_t)b * LSEQ + t) * DINNER + d] = (_Float16)sv;
}

struct ScanP {
  const float* dtw[3]; const float* dtb[3];
  const float* alog[3]; const float* Dp[3];
};

// Fused selective scan: grid (BATCH, 3 directions), 128 threads (one channel d
// per lane). State h[16] and A[d][:] live in VGPRs. The per-step 48-float
// x_dbl rows are streamed in CHUNK-row tiles by the Tensor Data Mover into a
// double-buffered LDS stage (wave 0 issues, s_wait_tensorcnt + one barrier
// per chunk), overlapping the DMA with the exp-chain compute.
__global__ void scan_fused(const float* __restrict__ xc32, const float* __restrict__ xdbl,
                           ScanP P, float* __restrict__ ytot) {
  int br = blockIdx.y;
  int b  = blockIdx.x;
  int d  = threadIdx.x;  // 128
  __shared__ float sb[2][CHUNK * NPAD];

  const float* xdb  = xdbl + ((size_t)br * BL + (size_t)b * LSEQ) * NPAD;
  const float* urow = xc32 + ((size_t)br * BATCH + b) * DINNER * (size_t)LSEQ + (size_t)d * LSEQ;
  float*       yrow = ytot + ((size_t)br * BATCH + b) * DINNER * (size_t)LSEQ + (size_t)d * LSEQ;

  float A[DSTATE];
  #pragma unroll
  for (int n = 0; n < DSTATE; ++n) A[n] = -__expf(P.alog[br][d * DSTATE + n]);
  float w0 = P.dtw[br][d * 4 + 0], w1 = P.dtw[br][d * 4 + 1];
  float w2 = P.dtw[br][d * 4 + 2], w3 = P.dtw[br][d * 4 + 3];
  float bias = P.dtb[br][d], Dd = P.Dp[br][d];
  float h[DSTATE];
  #pragma unroll
  for (int n = 0; n < DSTATE; ++n) h[n] = 0.f;

  if (threadIdx.x < 32) {               // wave 0 drives the TDM
    tdm_load_rows(xdb, &sb[0][0], CHUNK);
    __builtin_amdgcn_s_wait_tensorcnt(0);
  }
  __syncthreads();

  const int NCH = LSEQ / CHUNK;
  for (int c = 0; c < NCH; ++c) {
    int p = c & 1;
    if (threadIdx.x < 32 && (c + 1) < NCH)
      tdm_load_rows(xdb + (size_t)(c + 1) * CHUNK * NPAD, &sb[1 - p][0], CHUNK);

    const float4* up = reinterpret_cast<const float4*>(urow + c * CHUNK);
    float4 ua = up[0], ub = up[1];
    float uv[CHUNK] = {ua.x, ua.y, ua.z, ua.w, ub.x, ub.y, ub.z, ub.w};

    #pragma unroll
    for (int i = 0; i < CHUNK; ++i) {
      const float* row = &sb[p][i * NPAD];
      float dtv = bias + w0 * row[0] + w1 * row[1] + w2 * row[2] + w3 * row[3];
      float delta = (dtv > 20.f) ? dtv : log1pf(__expf(dtv));  // softplus
      float du = delta * uv[i];
      float y = 0.f;
      #pragma unroll
      for (int n = 0; n < DSTATE; ++n) {
        h[n] = __expf(delta * A[n]) * h[n] + du * row[4 + n];
        y += h[n] * row[20 + n];
      }
      yrow[smap(c * CHUNK + i, br)] = y + Dd * uv[i];
    }

    if (threadIdx.x < 32) __builtin_amdgcn_s_wait_tensorcnt(0);
    __syncthreads();
  }
}

// (y_fwd + y_bwd + y_slc) * silu(z) -> f16 (BL x 128) for out_proj
__global__ void combine_silu(const float* __restrict__ ytot, const _Float16* __restrict__ xz,
                             _Float16* __restrict__ yh) {
  int idx = blockIdx.x * blockDim.x + threadIdx.x;
  if (idx >= BATCH * DINNER * LSEQ) return;
  int t = idx & (LSEQ - 1);
  int d = (idx >> 12) & (DINNER - 1);
  int b = idx >> 19;
  size_t bi = ((size_t)b * DINNER + d) * LSEQ + t;
  const size_t BR = (size_t)BATCH * DINNER * LSEQ;
  float y = ytot[bi] + ytot[BR + bi] + ytot[2 * BR + bi];
  float z = (float)xz[((size_t)b * 256 + DINNER + d) * LSEQ + t];
  yh[((size_t)b * LSEQ + t) * DINNER + d] = (_Float16)(y * silu(z));
}

// 3x3 depthwise conv (pad 1) + GELU, out f16 (BL x 256)
__global__ void dwconv_gelu(const float* __restrict__ fin, const float* __restrict__ wdw,
                            _Float16* __restrict__ outh) {
  int idx = blockIdx.x * blockDim.x + threadIdx.x;  // B*256*4096
  if (idx >= BATCH * 256 * LSEQ) return;
  int l = idx & (LSEQ - 1);
  int oc = (idx >> 12) & 255;
  int b = idx >> 20;
  int wq = l & 63, hq = l >> 6;
  const float* base = fin + ((size_t)b * 256 + oc) * LSEQ;
  const float* wk = wdw + oc * 9;
  float acc = 0.f;
  #pragma unroll
  for (int dy = 0; dy < 3; ++dy) {
    int yy = hq + dy - 1;
    if (yy < 0 || yy >= 64) continue;
    #pragma unroll
    for (int dx = 0; dx < 3; ++dx) {
      int xx = wq + dx - 1;
      if (xx < 0 || xx >= 64) continue;
      acc += base[yy * 64 + xx] * wk[dy * 3 + dx];
    }
  }
  outh[((size_t)b * LSEQ + l) * 256 + oc] = (_Float16)gelu_exact(acc);
}

// ---------------------------------------------------------------- launch
extern "C" void kernel_launch(void* const* d_in, const int* in_sizes, int n_in,
                              void* d_out, int out_size, void* d_ws, size_t ws_size,
                              hipStream_t stream) {
  if (n_in < 31) return;
  const float* x_in = (const float*)d_in[0];
  const float* ln1g = (const float*)d_in[1];
  const float* ln1b = (const float*)d_in[2];
  const float* ln2g = (const float*)d_in[3];
  const float* ln2b = (const float*)d_in[4];
  const float* inpw = (const float*)d_in[5];   // (256,64)
  const float* outw = (const float*)d_in[6];   // (64,128)
  const float* ffw1 = (const float*)d_in[28];  // (256,64)
  const float* ffdw = (const float*)d_in[29];  // (256,1,3,3)
  const float* ffw2 = (const float*)d_in[30];  // (64,256)

  char* ws = (char*)d_ws;
  size_t off = 0;
  auto alloc = [&](size_t bytes) { size_t o = off; off = (off + bytes + 255) & ~(size_t)255; return o; };
  _Float16* th    = (_Float16*)(ws + alloc((size_t)BL * 64 * 2));
  _Float16* xz    = (_Float16*)(ws + alloc((size_t)BATCH * 256 * LSEQ * 2));
  float*    xc32  = (float*)   (ws + alloc((size_t)3 * BATCH * DINNER * LSEQ * 4)); // per-branch u
  _Float16* xc16  = (_Float16*)(ws + alloc((size_t)BL * DINNER * 2));               // reused
  float*    xdbl  = (float*)   (ws + alloc((size_t)3 * BL * NPAD * 4));             // per-branch
  float*    ytot  = (float*)   (ws + alloc((size_t)3 * BATCH * DINNER * LSEQ * 4)); // per-branch
  _Float16* yh    = (_Float16*)(ws + alloc((size_t)BL * DINNER * 2));
  float*    h2    = (float*)   (ws + alloc((size_t)BL * 64 * 4));
  _Float16* ln2h  = (_Float16*)(ws + alloc((size_t)BL * 64 * 2));
  float*    ffh   = (float*)   (ws + alloc((size_t)BL * 256 * 4));
  _Float16* ffh16 = (_Float16*)(ws + alloc((size_t)BL * 256 * 2));
  _Float16* w_inp = (_Float16*)(ws + alloc((size_t)256 * 64 * 2));
  _Float16* w_xp  = (_Float16*)(ws + alloc((size_t)3 * NPAD * DINNER * 2));
  _Float16* w_out = (_Float16*)(ws + alloc((size_t)64 * DINNER * 2));
  _Float16* w_ff1 = (_Float16*)(ws + alloc((size_t)256 * 64 * 2));
  _Float16* w_ff2 = (_Float16*)(ws + alloc((size_t)64 * 256 * 2));
  if (off > ws_size) return;

  auto cvt = [&](const float* s, _Float16* d, int n) {
    cvt_f16<<<(n + 255) / 256, 256, 0, stream>>>(s, d, n);
  };

  // weight prep
  zero_f16<<<(3 * NPAD * DINNER + 255) / 256, 256, 0, stream>>>(w_xp, 3 * NPAD * DINNER);
  cvt(inpw, w_inp, 256 * 64);
  for (int br = 0; br < 3; ++br)
    cvt((const float*)d_in[7 + 7 * br + 2], w_xp + (size_t)br * NPAD * DINNER, 36 * DINNER);
  cvt(outw, w_out, 64 * DINNER);
  cvt(ffw1, w_ff1, 256 * 64);
  cvt(ffw2, w_ff2, 64 * 256);

  // LN1 + in_proj
  ln_nchw<<<BL / 8, 256, 0, stream>>>(x_in, ln1g, ln1b, th);
  gemm_wmma<0, 4><<<(BL / 16) * (256 / 16) / 4 / 4, 128, 0, stream>>>(
      th, w_inp, BL, 256, 64, nullptr, nullptr, xz);

  // per-branch conv + x_proj (xc16 reused; xc32/xdbl kept per-branch)
  const size_t BRU = (size_t)BATCH * DINNER * LSEQ;
  int nElem = BATCH * DINNER * LSEQ;
  for (int br = 0; br < 3; ++br) {
    conv_silu<<<(nElem + 255) / 256, 256, 0, stream>>>(
        xz, (const float*)d_in[7 + 7 * br + 0], (const float*)d_in[7 + 7 * br + 1],
        xc32 + br * BRU, xc16, br);
    gemm_wmma<1, 3><<<(BL / 16) * (NPAD / 16) / 3 / 4, 128, 0, stream>>>(
        xc16, w_xp + (size_t)br * NPAD * DINNER, BL, NPAD, DINNER,
        xdbl + (size_t)br * BL * NPAD, nullptr, nullptr);
  }

  // one fused scan over all (batch, direction) pairs
  ScanP sp;
  for (int br = 0; br < 3; ++br) {
    sp.dtw[br]  = (const float*)d_in[7 + 7 * br + 3];
    sp.dtb[br]  = (const float*)d_in[7 + 7 * br + 4];
    sp.alog[br] = (const float*)d_in[7 + 7 * br + 5];
    sp.Dp[br]   = (const float*)d_in[7 + 7 * br + 6];
  }
  scan_fused<<<dim3(BATCH, 3), DINNER, 0, stream>>>(xc32, xdbl, sp, ytot);

  // gate + out_proj + residual
  combine_silu<<<(nElem + 255) / 256, 256, 0, stream>>>(ytot, xz, yh);
  gemm_wmma<2, 4><<<(BL / 16) * (64 / 16) / 4 / 4, 128, 0, stream>>>(
      yh, w_out, BL, 64, DINNER, h2, x_in, nullptr);

  // LN2 + FF
  ln_rowmajor<<<BL / 8, 256, 0, stream>>>(h2, ln2g, ln2b, ln2h);
  gemm_wmma<3, 4><<<(BL / 16) * (256 / 16) / 4 / 4, 128, 0, stream>>>(
      ln2h, w_ff1, BL, 256, 64, ffh, nullptr, nullptr);
  dwconv_gelu<<<(BATCH * 256 * LSEQ + 255) / 256, 256, 0, stream>>>(ffh, ffdw, ffh16);
  gemm_wmma<4, 4><<<(BL / 16) * (64 / 16) / 4 / 4, 128, 0, stream>>>(
      ffh16, w_ff2, BL, 64, 256, (float*)d_out, h2, nullptr);
}
